// GumbelSigmoidMask_40621800685937
// MI455X (gfx1250) — matrix operand order repaired and checked by reading, not compiled, not run on hardware
//
#include <hip/hip_runtime.h>
#include <hip/hip_bf16.h>

typedef __attribute__((ext_vector_type(16))) _Float16 v16h;
typedef __attribute__((ext_vector_type(8)))  _Float16 v8h;
typedef __attribute__((ext_vector_type(8)))  float    v8f;

#define GS_EPS 1e-8f

__device__ __forceinline__ float gumbel_sigmoid(float logit, float u) {
    u = fminf(fmaxf(u, GS_EPS), 1.0f - GS_EPS);
    float noise = __logf(u) - log1pf(-u);
    float z = logit + noise;          // TEMPERATURE = 1.0
    return 1.0f / (1.0f + __expf(-z));
}

// ---------------------------------------------------------------------------
// Prep: swizzle + f16-convert the three 128x64 first-layer weight blocks
// (w1n, w1e[0:128], w1e[128:256]) into WMMA B-operand layout.
// Tile t = ((w*4 + ntile)*4 + kc); per lane a v16h: element e holds
// W(kc*32 + (lane>=16)*8 + e + (e>=8)*8,  ntile*16 + (lane&15)).
// ---------------------------------------------------------------------------
__global__ __launch_bounds__(32) void prep_kernel(const float* __restrict__ w1n,
                                                  const float* __restrict__ w1e,
                                                  v16h* __restrict__ Bpack) {
    int t = blockIdx.x;          // 0..47
    int lane = threadIdx.x;      // 0..31
    int kc = t & 3;
    int nt = (t >> 2) & 3;
    int w  = t >> 4;             // 0: w1n, 1: w1e_lo, 2: w1e_hi
    int col = nt * 16 + (lane & 15);
    int kbase = kc * 32 + (lane >> 4) * 8;
    v16h b;
#pragma unroll
    for (int e = 0; e < 16; ++e) {
        int k = kbase + e + (e >= 8 ? 8 : 0);
        float v;
        if (w == 0)      v = w1n[k * 64 + col];
        else if (w == 1) v = w1e[k * 64 + col];
        else             v = w1e[(k + 128) * 64 + col];
        b[e] = (_Float16)v;
    }
    Bpack[(size_t)t * 32 + lane] = b;
}

// ---------------------------------------------------------------------------
// Node kernel: one wave per 16-row tile of x.
//   - A tiles (16x32 f16) built from x per WMMA A layout
//   - 48x v_wmma_f32_16x16x32_f16: H_node(16x64), P1(16x64), P2(16x64)
//   - P1(+b1e baked) and P2 stored as f16 for the edge kernel
//   - node second layer 64->1 via in-register relu-dot + shfl_xor reduction
//   - tile id / tail predicate forced scalar via readfirstlane so the
//     full-tile fast path is an s_cbranch, not an exec-mask dance
// ---------------------------------------------------------------------------
__global__ __launch_bounds__(128) void node_kernel(
    const float* __restrict__ x, const float* __restrict__ u_node,
    const float* __restrict__ b1n, const float* __restrict__ w2n,
    const float* __restrict__ b2n, const float* __restrict__ b1e,
    const v16h* __restrict__ Bpack,
    _Float16* __restrict__ P1h, _Float16* __restrict__ P2h,
    float* __restrict__ out_node, int N, int ntiles)
{
    int lane = threadIdx.x & 31;
    // tile is identical across the wave; pin it to an SGPR
    int tile = __builtin_amdgcn_readfirstlane(blockIdx.x * 4 + (threadIdx.x >> 5));
    if (tile >= ntiles) return;            // scalar branch; EXEC stays all-ones
    int rowbase = tile << 4;
    int m  = lane & 15;
    int hh = lane >> 4;
    // scalar tail predicate (always 1 when 16 | N)
    int full = __builtin_amdgcn_readfirstlane((rowbase + 16 <= N) ? 1 : 0);

    int arow = rowbase + m;
    if (arow >= N) arow = N - 1;           // clamp for partial tail tile
    const float* xr = x + (size_t)arow * 128;

    // A tiles: per ISA 16-bit A layout (lanes 0-15: K 0-7/16-23, lanes 16-31: +8)
    v16h a[4];
#pragma unroll
    for (int kc = 0; kc < 4; ++kc) {
        int k0 = kc * 32 + hh * 8;
        float4 f0 = *(const float4*)(xr + k0);
        float4 f1 = *(const float4*)(xr + k0 + 4);
        float4 f2 = *(const float4*)(xr + k0 + 16);
        float4 f3 = *(const float4*)(xr + k0 + 20);
        a[kc][0]  = (_Float16)f0.x; a[kc][1]  = (_Float16)f0.y;
        a[kc][2]  = (_Float16)f0.z; a[kc][3]  = (_Float16)f0.w;
        a[kc][4]  = (_Float16)f1.x; a[kc][5]  = (_Float16)f1.y;
        a[kc][6]  = (_Float16)f1.z; a[kc][7]  = (_Float16)f1.w;
        a[kc][8]  = (_Float16)f2.x; a[kc][9]  = (_Float16)f2.y;
        a[kc][10] = (_Float16)f2.z; a[kc][11] = (_Float16)f2.w;
        a[kc][12] = (_Float16)f3.x; a[kc][13] = (_Float16)f3.y;
        a[kc][14] = (_Float16)f3.z; a[kc][15] = (_Float16)f3.w;
    }

    // ---- P1 (w=1, bias b1e baked in) and P2 (w=2): compute + store f16 ----
#pragma unroll
    for (int w = 1; w <= 2; ++w) {
#pragma unroll
        for (int nt = 0; nt < 4; ++nt) {
            v8f c = {};
#pragma unroll
            for (int kc = 0; kc < 4; ++kc) {
                v16h b = Bpack[(size_t)(((w * 4 + nt) * 4 + kc) * 32 + lane)];
                c = __builtin_amdgcn_wmma_f32_16x16x32_f16(
                        false, a[kc], false, b, (short)0, c, false, false);
            }
            int col = nt * 16 + m;
            float bias = (w == 1) ? b1e[col] : 0.0f;
            _Float16* dst = (w == 1) ? P1h : P2h;
            // C/D layout: element r -> row = rowbase + 8*(lane>=16) + r
            _Float16* dp = dst + (size_t)(rowbase + hh * 8) * 64 + col;
            if (full) {
#pragma unroll
                for (int r = 0; r < 8; ++r)
                    dp[(size_t)r * 64] = (_Float16)(c[r] + bias);
            } else {
#pragma unroll
                for (int r = 0; r < 8; ++r) {
                    int row = rowbase + hh * 8 + r;
                    if (row < N) dp[(size_t)r * 64] = (_Float16)(c[r] + bias);
                }
            }
        }
    }

    // ---- node hidden layer ----
    v8f cn[4];
#pragma unroll
    for (int nt = 0; nt < 4; ++nt) {
        v8f c = {};
#pragma unroll
        for (int kc = 0; kc < 4; ++kc) {
            v16h b = Bpack[(size_t)((nt * 4 + kc) * 32 + lane)];
            c = __builtin_amdgcn_wmma_f32_16x16x32_f16(
                    false, a[kc], false, b, (short)0, c, false, false);
        }
        cn[nt] = c;
    }

    // ---- node second layer: relu(h+b1n) . w2n, reduced across 16 lanes ----
    float b1v[4], w2v[4];
#pragma unroll
    for (int nt = 0; nt < 4; ++nt) {
        int h = nt * 16 + m;
        b1v[nt] = b1n[h];
        w2v[nt] = w2n[h];
    }
    float b2 = b2n[0];

    float pr[8];
#pragma unroll
    for (int r = 0; r < 8; ++r) {
        float p = 0.0f;
#pragma unroll
        for (int nt = 0; nt < 4; ++nt) {
            float v = cn[nt][r] + b1v[nt];
            v = v > 0.0f ? v : 0.0f;
            p += v * w2v[nt];
        }
        // reduce within each 16-lane half (xor of bits 0..3 stays in-half)
        p += __shfl_xor(p, 1, 32);
        p += __shfl_xor(p, 2, 32);
        p += __shfl_xor(p, 4, 32);
        p += __shfl_xor(p, 8, 32);
        pr[r] = p;
    }
    if (m == 0) {
        int row0 = rowbase + hh * 8;
        if (full) {
#pragma unroll
            for (int r = 0; r < 8; ++r)
                out_node[row0 + r] = gumbel_sigmoid(pr[r] + b2, u_node[row0 + r]);
        } else {
#pragma unroll
            for (int r = 0; r < 8; ++r)
                if (row0 + r < N)
                    out_node[row0 + r] = gumbel_sigmoid(pr[r] + b2, u_node[row0 + r]);
        }
    }
}

// ---------------------------------------------------------------------------
// Edge kernel: one thread per edge. Gather P1[src], P2[dst] (f16, L2-resident),
// packed-f16 relu-sum, f32 dot with w2e (staged in LDS), gumbel-sigmoid.
// ---------------------------------------------------------------------------
__global__ __launch_bounds__(256) void edge_kernel(
    const long long* __restrict__ ei, const float* __restrict__ u_edge,
    const float* __restrict__ w2e, const float* __restrict__ b2e,
    const _Float16* __restrict__ P1h, const _Float16* __restrict__ P2h,
    float* __restrict__ out_edge, int E)
{
    __shared__ float w2s[64];
    __shared__ float b2s;
    if (threadIdx.x < 64) w2s[threadIdx.x] = w2e[threadIdx.x];
    if (threadIdx.x == 64) b2s = b2e[0];
    __syncthreads();

    int e = blockIdx.x * blockDim.x + threadIdx.x;
    if (e >= E) return;

    int src = (int)ei[e];
    int dst = (int)ei[(size_t)E + e];
    const v8h* p1 = (const v8h*)(P1h + (size_t)src * 64);
    const v8h* p2 = (const v8h*)(P2h + (size_t)dst * 64);

    const v8h vzero = {};
    float acc = 0.0f;
#pragma unroll
    for (int j = 0; j < 8; ++j) {
        v8h s = p1[j] + p2[j];                       // v_pk_add_f16 (b1e baked in P1)
        s = __builtin_elementwise_max(s, vzero);     // v_pk_max relu
#pragma unroll
        for (int t = 0; t < 8; ++t)
            acc += (float)s[t] * w2s[j * 8 + t];
    }
    out_edge[e] = gumbel_sigmoid(acc + b2s, u_edge[e]);
}

// ---------------------------------------------------------------------------
extern "C" void kernel_launch(void* const* d_in, const int* in_sizes, int n_in,
                              void* d_out, int out_size, void* d_ws, size_t ws_size,
                              hipStream_t stream) {
    (void)n_in; (void)out_size; (void)ws_size;

    const float*     x      = (const float*)d_in[0];
    const long long* ei     = (const long long*)d_in[1];   // int64 [2, E]
    const float*     u_node = (const float*)d_in[2];
    const float*     u_edge = (const float*)d_in[3];
    const float*     w1n    = (const float*)d_in[4];
    const float*     b1n    = (const float*)d_in[5];
    const float*     w2n    = (const float*)d_in[6];
    const float*     b2n    = (const float*)d_in[7];
    const float*     w1e    = (const float*)d_in[8];
    const float*     b1e    = (const float*)d_in[9];
    const float*     w2e    = (const float*)d_in[10];
    const float*     b2e    = (const float*)d_in[11];

    int N = in_sizes[2];   // u_node length = n_graph_nodes
    int E = in_sizes[3];   // u_edge length = n_edges

    char* ws = (char*)d_ws;
    v16h* Bpack = (v16h*)ws;                               // 48*32*32B = 48KB
    size_t p1_off   = 65536;
    size_t p1_bytes = (size_t)N * 64 * sizeof(_Float16);
    size_t p2_off   = (p1_off + p1_bytes + 255) & ~(size_t)255;
    _Float16* P1h = (_Float16*)(ws + p1_off);
    _Float16* P2h = (_Float16*)(ws + p2_off);

    float* out_node = (float*)d_out;
    float* out_edge = out_node + N;

    prep_kernel<<<48, 32, 0, stream>>>(w1n, w1e, Bpack);

    int ntiles  = (N + 15) / 16;
    int nblocks = (ntiles + 3) / 4;
    node_kernel<<<nblocks, 128, 0, stream>>>(x, u_node, b1n, w2n, b2n, b1e,
                                             Bpack, P1h, P2h, out_node, N, ntiles);

    edge_kernel<<<(E + 255) / 256, 256, 0, stream>>>(ei, u_edge, w2e, b2e,
                                                     P1h, P2h, out_edge, E);
}
